// TI_TransformerNoEdges_23244363006446
// MI455X (gfx1250) — matrix operand order repaired and checked by reading, not compiled, not run on hardware
//
#include <hip/hip_runtime.h>
#include <math.h>

// ---------------------------------------------------------------------------
// CDNA5 (gfx1250) wave32 WMMA types
// ---------------------------------------------------------------------------
typedef _Float16 v16h __attribute__((ext_vector_type(16)));
typedef _Float16 h8   __attribute__((ext_vector_type(8)));
typedef float    v8f  __attribute__((ext_vector_type(8)));

#define NNODES 4096
#define DMODEL 256
#define NHEADS 4
#define HDIM   64
#define FFDIM  1024
#define NEDGE  65536
#define ETOT   (NEDGE + NNODES)
#define NGRAPH 32
#define OUTDIM 118

// ---------------------------------------------------------------------------
// small helpers
// ---------------------------------------------------------------------------
__device__ __forceinline__ unsigned fenc(float f) {
    unsigned i = __float_as_uint(f);
    return (i & 0x80000000u) ? ~i : (i | 0x80000000u);   // monotone float->u32
}
__device__ __forceinline__ float fdec(unsigned u) {
    unsigned b = (u & 0x80000000u) ? (u ^ 0x80000000u) : ~u;
    return __uint_as_float(b);
}

// ---------------------------------------------------------------------------
// feature_fc : h0[n,j] = sum_k x[n,k] W[k,j] + b[j]   (K=16, tiny -> scalar)
// ---------------------------------------------------------------------------
__global__ void k_feat_fc(const float* __restrict__ x, const float* __restrict__ W,
                          const float* __restrict__ b, float* __restrict__ h0) {
    int idx = blockIdx.x * blockDim.x + threadIdx.x;
    if (idx >= NNODES * 64) return;
    int n = idx >> 6, j = idx & 63;
    float s = b[j];
    #pragma unroll
    for (int k = 0; k < 16; ++k) s += x[n * 16 + k] * W[k * 64 + j];
    h0[idx] = s;
}

// ---------------------------------------------------------------------------
// fp32 -> fp16 cast (plain) and cast+transpose (for weights: [R][C] -> [C][R])
// ---------------------------------------------------------------------------
__global__ void k_cast_f16(const float* __restrict__ in, _Float16* __restrict__ out, int n) {
    int idx = blockIdx.x * blockDim.x + threadIdx.x;
    if (idx < n) out[idx] = (_Float16)in[idx];
}

__global__ void k_cast_f16_tr(const float* __restrict__ in, _Float16* __restrict__ out,
                              int R, int C) {
    int idx = blockIdx.x * blockDim.x + threadIdx.x;
    if (idx >= R * C) return;
    int r = idx / C, c = idx % C;
    out[(size_t)c * R + r] = (_Float16)in[idx];
}

// ---------------------------------------------------------------------------
// WMMA GEMM:  C[M,N] = A[M,K](f16,row-major) x B[K,N] given as BT[N][K] (f16)
// one wave -> 32x64 tile, 8x v_wmma_f32_16x16x32_f16 per 32-wide k-step.
// All fragment loads are contiguous 2x b128 per lane.
// Epilogue: f32 out (Cf), f16 out (Ch), or f16 transposed out (ChT, [N][M]).
// ---------------------------------------------------------------------------
__global__ void gemm_wmma_f16(const _Float16* __restrict__ A, const _Float16* __restrict__ BT,
                              const float* __restrict__ bias, float* __restrict__ Cf,
                              _Float16* __restrict__ Ch, _Float16* __restrict__ ChT,
                              int M, int N, int K, int lda, int ldc, int relu) {
    int wave   = threadIdx.x >> 5;
    int tilesN = N >> 6;
    int tiles  = (M >> 5) * tilesN;
    int tile   = blockIdx.x * (blockDim.x >> 5) + wave;
    if (tile >= tiles) return;
    int m0 = (tile / tilesN) << 5;
    int n0 = (tile % tilesN) << 6;
    int lane = threadIdx.x & 31;
    int hs = lane >> 4;           // half-wave select
    int lm = lane & 15;

    v8f acc[2][4] = {};
    for (int kk = 0; kk < K; kk += 32) {
        // two A fragments (rows m0.. and m0+16..), contiguous 2x b128 each
        v16h af[2];
        #pragma unroll
        for (int u = 0; u < 2; ++u) {
            const _Float16* ar = A + (size_t)(m0 + u * 16 + lm) * lda + kk + hs * 8;
            h8 lo = *(const h8*)ar;
            h8 hi = *(const h8*)(ar + 16);
            #pragma unroll
            for (int i = 0; i < 8; ++i) { af[u][i] = lo[i]; af[u][i + 8] = hi[i]; }
        }
        #pragma unroll
        for (int t = 0; t < 4; ++t) {
            // B fragment from transposed weights: contiguous 2x b128
            int col = n0 + t * 16 + lm;
            const _Float16* bp = BT + (size_t)col * K + kk + hs * 8;
            h8 lo = *(const h8*)bp;
            h8 hi = *(const h8*)(bp + 16);
            v16h bf;
            #pragma unroll
            for (int i = 0; i < 8; ++i) { bf[i] = lo[i]; bf[i + 8] = hi[i]; }
            #pragma unroll
            for (int u = 0; u < 2; ++u)
                acc[u][t] = __builtin_amdgcn_wmma_f32_16x16x32_f16(
                    false, af[u], false, bf, (short)0, acc[u][t], false, false);
        }
    }
    #pragma unroll
    for (int u = 0; u < 2; ++u) {
        #pragma unroll
        for (int t = 0; t < 4; ++t) {
            int col = n0 + t * 16 + lm;
            float bv = bias ? bias[col] : 0.0f;
            #pragma unroll
            for (int r = 0; r < 8; ++r) {
                int row = m0 + u * 16 + r + hs * 8;
                float v = acc[u][t][r] + bv;
                if (relu) v = fmaxf(v, 0.0f);
                if (Cf)  Cf[(size_t)row * ldc + col] = v;
                if (Ch)  Ch[(size_t)row * ldc + col] = (_Float16)v;
                if (ChT) ChT[(size_t)col * M + row] = (_Float16)v;
            }
        }
    }
}

// ---------------------------------------------------------------------------
// GAT: per-node attention logits  a_s[n,h]=<hW[n,h,:],a_src[h,:]>, same a_d
// ---------------------------------------------------------------------------
__global__ void k_node_attn(const float* __restrict__ hW, const float* __restrict__ a_src,
                            const float* __restrict__ a_dst, float* __restrict__ as,
                            float* __restrict__ ad) {
    int idx = blockIdx.x * blockDim.x + threadIdx.x;
    if (idx >= NNODES * NHEADS) return;
    int n = idx >> 2, h = idx & 3;
    float s0 = 0.f, s1 = 0.f;
    const float* row = hW + (size_t)n * DMODEL + h * HDIM;
    #pragma unroll 8
    for (int k = 0; k < HDIM; ++k) {
        float v = row[k];
        s0 += v * a_src[h * HDIM + k];
        s1 += v * a_dst[h * HDIM + k];
    }
    as[idx] = s0; ad[idx] = s1;
}

__global__ void k_gat_init(float* __restrict__ C, unsigned* __restrict__ mseg,
                           float* __restrict__ sseg) {
    int idx = blockIdx.x * blockDim.x + threadIdx.x;
    if (idx < NNODES * DMODEL) C[idx] = 0.0f;
    if (idx < NNODES * NHEADS) { mseg[idx] = 0u; sseg[idx] = 0.0f; }
}

__global__ void k_gat_edge1(const int* __restrict__ ei, const float* __restrict__ as,
                            const float* __restrict__ ad, float* __restrict__ ew,
                            unsigned* __restrict__ mseg) {
    int idx = blockIdx.x * blockDim.x + threadIdx.x;
    if (idx >= ETOT * NHEADS) return;
    int e = idx >> 2, h = idx & 3;
    int src = (e < NEDGE) ? ei[e] : (e - NEDGE);
    int dst = (e < NEDGE) ? ei[NEDGE + e] : (e - NEDGE);
    float v = as[src * NHEADS + h] + ad[dst * NHEADS + h];
    v = (v > 0.0f) ? v : 0.2f * v;                        // leaky relu
    ew[idx] = v;
    atomicMax(&mseg[dst * NHEADS + h], fenc(v));
}

__global__ void k_gat_edge2(const int* __restrict__ ei, float* __restrict__ ew,
                            const unsigned* __restrict__ mseg, float* __restrict__ sseg) {
    int idx = blockIdx.x * blockDim.x + threadIdx.x;
    if (idx >= ETOT * NHEADS) return;
    int e = idx >> 2, h = idx & 3;
    int dst = (e < NEDGE) ? ei[NEDGE + e] : (e - NEDGE);
    float w = __expf(ew[idx] - fdec(mseg[dst * NHEADS + h]));
    ew[idx] = w;
    atomicAdd(&sseg[dst * NHEADS + h], w);
}

__global__ void k_gat_edge3(const int* __restrict__ ei, const float* __restrict__ ew,
                            const float* __restrict__ sseg, const float* __restrict__ hW,
                            float* __restrict__ C) {
    int e = blockIdx.x, c = threadIdx.x;            // 256 channels
    int h = c >> 6;
    int src = (e < NEDGE) ? ei[e] : (e - NEDGE);
    int dst = (e < NEDGE) ? ei[NEDGE + e] : (e - NEDGE);
    float alpha = ew[e * NHEADS + h] / (sseg[dst * NHEADS + h] + 1e-16f);
    atomicAdd(&C[(size_t)dst * DMODEL + c], alpha * hW[(size_t)src * DMODEL + c]);
}

__global__ void k_gat_final(const float* __restrict__ C, const float* __restrict__ b,
                            float* __restrict__ out) {
    int idx = blockIdx.x * blockDim.x + threadIdx.x;
    if (idx >= NNODES * DMODEL) return;
    out[idx] = fmaxf(C[idx] + b[idx & (DMODEL - 1)], 0.0f);
}

// ---------------------------------------------------------------------------
// Flash attention (one wave == one 16-query block of one head), WMMA QK^T / PV
// online softmax; P re-layout C->A via LDS. V is given transposed: VT[D][N].
// All global fragment loads are contiguous 2x b128; LDS reads are b128.
// ---------------------------------------------------------------------------
__global__ void flash_attn_wmma(const _Float16* __restrict__ Q, const _Float16* __restrict__ Kx,
                                const _Float16* __restrict__ VT, float* __restrict__ Out,
                                int N, int D) {
    int q0  = blockIdx.x * 16;
    int ch0 = blockIdx.y * HDIM;
    int lane = threadIdx.x & 31;
    int hs = lane >> 4, lm = lane & 15;
    __shared__ _Float16 sP[16 * 32];

    // Q A-fragments for the two 32-wide hd chunks
    v16h qa[2];
    #pragma unroll
    for (int c = 0; c < 2; ++c) {
        const _Float16* qr = Q + (size_t)(q0 + lm) * D + ch0 + c * 32 + hs * 8;
        h8 lo = *(const h8*)qr;
        h8 hi = *(const h8*)(qr + 16);
        #pragma unroll
        for (int i = 0; i < 8; ++i) { qa[c][i] = lo[i]; qa[c][i + 8] = hi[i]; }
    }

    float mrun[8], lrun[8];
    v8f oacc[4] = {};
    #pragma unroll
    for (int r = 0; r < 8; ++r) { mrun[r] = -1e30f; lrun[r] = 0.0f; }
    const float scale = 0.125f;   // 1/sqrt(64)

    for (int j = 0; j < N; j += 32) {
        // two 16x16 score tiles (keys j..j+15, j+16..j+31)
        v8f sc[2] = {};
        #pragma unroll
        for (int s = 0; s < 2; ++s) {
            #pragma unroll
            for (int c = 0; c < 2; ++c) {
                // K^T B-fragment: two 8-half runs (hs*8, hs*8+16) of one key row
                const _Float16* kr = Kx + (size_t)(j + s * 16 + lm) * D + ch0 + c * 32 + hs * 8;
                h8 lo = *(const h8*)kr;
                h8 hi = *(const h8*)(kr + 16);
                v16h bf;
                #pragma unroll
                for (int i = 0; i < 8; ++i) { bf[i] = lo[i]; bf[i + 8] = hi[i]; }
                sc[s] = __builtin_amdgcn_wmma_f32_16x16x32_f16(
                    false, qa[c], false, bf, (short)0, sc[s], false, false);
            }
        }
        // online softmax per row (row lives across one 16-lane half-wave)
        #pragma unroll
        for (int r = 0; r < 8; ++r) {
            float a0 = sc[0][r] * scale, a1 = sc[1][r] * scale;
            float v = fmaxf(a0, a1);
            v = fmaxf(v, __shfl_xor(v, 1, 32));
            v = fmaxf(v, __shfl_xor(v, 2, 32));
            v = fmaxf(v, __shfl_xor(v, 4, 32));
            v = fmaxf(v, __shfl_xor(v, 8, 32));
            float mnew = fmaxf(mrun[r], v);
            float al = __expf(mrun[r] - mnew);
            float p0 = __expf(a0 - mnew), p1 = __expf(a1 - mnew);
            float rs = p0 + p1;
            rs += __shfl_xor(rs, 1, 32);
            rs += __shfl_xor(rs, 2, 32);
            rs += __shfl_xor(rs, 4, 32);
            rs += __shfl_xor(rs, 8, 32);
            lrun[r] = lrun[r] * al + rs;
            mrun[r] = mnew;
            sc[0][r] = p0; sc[1][r] = p1;
            #pragma unroll
            for (int t = 0; t < 4; ++t) oacc[t][r] *= al;
        }
        // C-layout -> row-major LDS -> A-layout for P
        __syncthreads();
        #pragma unroll
        for (int r = 0; r < 8; ++r) {
            sP[(r + hs * 8) * 32 + lm]      = (_Float16)sc[0][r];
            sP[(r + hs * 8) * 32 + 16 + lm] = (_Float16)sc[1][r];
        }
        __syncthreads();
        v16h pa;
        {
            const _Float16* pr = &sP[lm * 32 + hs * 8];
            h8 plo = *(const h8*)pr;            // ds_load_b128
            h8 phi = *(const h8*)(pr + 16);
            #pragma unroll
            for (int i = 0; i < 8; ++i) { pa[i] = plo[i]; pa[i + 8] = phi[i]; }
        }
        // P(16x32) x V(32x64); V fragments contiguous from VT
        #pragma unroll
        for (int t = 0; t < 4; ++t) {
            int col = ch0 + t * 16 + lm;
            const _Float16* vp = VT + (size_t)col * N + j + hs * 8;
            h8 lo = *(const h8*)vp;
            h8 hi = *(const h8*)(vp + 16);
            v16h bf;
            #pragma unroll
            for (int i = 0; i < 8; ++i) { bf[i] = lo[i]; bf[i + 8] = hi[i]; }
            oacc[t] = __builtin_amdgcn_wmma_f32_16x16x32_f16(
                false, pa, false, bf, (short)0, oacc[t], false, false);
        }
    }
    #pragma unroll
    for (int r = 0; r < 8; ++r) {
        float inv = 1.0f / lrun[r];
        int row = q0 + r + hs * 8;
        #pragma unroll
        for (int t = 0; t < 4; ++t)
            Out[(size_t)row * D + ch0 + t * 16 + lm] = oacc[t][r] * inv;
    }
}

// ---------------------------------------------------------------------------
// out = LayerNorm(x + y) * g + b    (one 256-thread block per row)
// ---------------------------------------------------------------------------
__global__ void k_residual_ln(const float* __restrict__ x, const float* __restrict__ y,
                              const float* __restrict__ g, const float* __restrict__ b,
                              float* __restrict__ out) {
    __shared__ float s1[DMODEL], s2[DMODEL];
    int row = blockIdx.x, tid = threadIdx.x;
    float v = x[(size_t)row * DMODEL + tid] + y[(size_t)row * DMODEL + tid];
    s1[tid] = v; s2[tid] = v * v;
    __syncthreads();
    for (int off = DMODEL / 2; off > 0; off >>= 1) {
        if (tid < off) { s1[tid] += s1[tid + off]; s2[tid] += s2[tid + off]; }
        __syncthreads();
    }
    float mean = s1[0] * (1.0f / DMODEL);
    float var  = s2[0] * (1.0f / DMODEL) - mean * mean;
    out[(size_t)row * DMODEL + tid] = (v - mean) * rsqrtf(var + 1e-5f) * g[tid] + b[tid];
}

// ---------------------------------------------------------------------------
// global max pool over contiguous 128-node graphs, then 2 small FCs
// ---------------------------------------------------------------------------
__global__ void k_pool_max(const float* __restrict__ h, float* __restrict__ pooled) {
    int idx = blockIdx.x * blockDim.x + threadIdx.x;
    if (idx >= NGRAPH * DMODEL) return;
    int g = idx >> 8, c = idx & (DMODEL - 1);
    float m = -1e30f;
    for (int i = 0; i < NNODES / NGRAPH; ++i)
        m = fmaxf(m, h[(size_t)(g * (NNODES / NGRAPH) + i) * DMODEL + c]);
    pooled[idx] = m;
}

__global__ void k_fc(const float* __restrict__ X, const float* __restrict__ W,
                     const float* __restrict__ b, float* __restrict__ Y,
                     int M, int N, int K, int relu) {
    int idx = blockIdx.x * blockDim.x + threadIdx.x;
    if (idx >= M * N) return;
    int m = idx / N, n = idx % N;
    float s = b[n];
    for (int k = 0; k < K; ++k) s += X[m * K + k] * W[k * N + n];
    Y[idx] = relu ? fmaxf(s, 0.0f) : s;
}

// ---------------------------------------------------------------------------
// host orchestration
// ---------------------------------------------------------------------------
extern "C" void kernel_launch(void* const* d_in, const int* in_sizes, int n_in,
                              void* d_out, int out_size, void* d_ws, size_t ws_size,
                              hipStream_t stream) {
    (void)in_sizes; (void)n_in; (void)out_size; (void)ws_size;
    auto F = [&](int i) { return (const float*)d_in[i]; };
    const float* x_in = F(0);
    const int*   ei   = (const int*)d_in[1];
    // d_in[2] = batch (contiguous groups; exploited directly)
    const float* featW = F(3); const float* featB = F(4);
    // gat l: 5+4l .. ; dec: sa 21-28, ca 29-36, ln 37-42, W1/b1/W2/b2 43-46
    const float* fc1W = F(47); const float* fc1b = F(48);
    const float* fc2W = F(49); const float* fc2b = F(50);

    // workspace layout (256B aligned)
    char* ws = (char*)d_ws;
    size_t cur = 0;
    auto take = [&](size_t bytes) { size_t o = cur; cur = (cur + bytes + 255) & ~(size_t)255; return o; };
    float* A    = (float*)(ws + take((size_t)NNODES * DMODEL * 4));   // h / x2
    float* Bm   = (float*)(ws + take((size_t)NNODES * DMODEL * 4));   // hW / proj out
    float* C    = (float*)(ws + take((size_t)NNODES * DMODEL * 4));   // aggregation / attn out
    float* X1   = (float*)(ws + take((size_t)NNODES * DMODEL * 4));
    float* AS   = (float*)(ws + take((size_t)NNODES * NHEADS * 4));
    float* AD   = (float*)(ws + take((size_t)NNODES * NHEADS * 4));
    float* SS   = (float*)(ws + take((size_t)NNODES * NHEADS * 4));
    unsigned* MS = (unsigned*)(ws + take((size_t)NNODES * NHEADS * 4));
    float* EW   = (float*)(ws + take((size_t)ETOT * NHEADS * 4));
    float* POOL = (float*)(ws + take((size_t)NGRAPH * DMODEL * 4));
    float* Y1   = (float*)(ws + take((size_t)NGRAPH * 128 * 4));
    _Float16* HH = (_Float16*)(ws + take((size_t)NNODES * DMODEL * 2));
    _Float16* H2 = (_Float16*)(ws + take((size_t)NNODES * DMODEL * 2));
    _Float16* QH = (_Float16*)(ws + take((size_t)NNODES * DMODEL * 2));
    _Float16* KH = (_Float16*)(ws + take((size_t)NNODES * DMODEL * 2));
    _Float16* VT = (_Float16*)(ws + take((size_t)NNODES * DMODEL * 2));  // transposed V
    _Float16* TH = (_Float16*)(ws + take((size_t)NNODES * FFDIM * 2));
    _Float16* WT = (_Float16*)(ws + take((size_t)DMODEL * FFDIM * 2));   // transposed weights

    auto cast16 = [&](const float* src, _Float16* dst, int n) {
        k_cast_f16<<<(n + 255) / 256, 256, 0, stream>>>(src, dst, n);
    };
    auto castW = [&](const float* src, int R, int Cc) {   // [R][Cc] f32 -> WT[Cc][R] f16
        k_cast_f16_tr<<<(R * Cc + 255) / 256, 256, 0, stream>>>(src, WT, R, Cc);
    };
    auto gemm = [&](const _Float16* Ain, const float* bias,
                    float* Cf, _Float16* Ch, _Float16* ChT, int M, int N, int K,
                    int lda, int ldc, int relu) {
        int tiles = (M / 32) * (N / 64);
        gemm_wmma_f16<<<(tiles + 7) / 8, 256, 0, stream>>>(Ain, WT, bias, Cf, Ch, ChT,
                                                           M, N, K, lda, ldc, relu);
    };
    const int TPB = 256;
    int nED = ETOT * NHEADS;

    // feature_fc -> A (N x 64)
    k_feat_fc<<<(NNODES * 64 + TPB - 1) / TPB, TPB, 0, stream>>>(x_in, featW, featB, A);

    // ---- 4 GAT layers ----
    for (int l = 0; l < 4; ++l) {
        int din = (l == 0) ? 64 : DMODEL;
        int base = 5 + l * 4;
        const float* Wg = F(base), *asr = F(base + 1), *adr = F(base + 2), *bg = F(base + 3);
        cast16(A, HH, NNODES * din);
        castW(Wg, din, DMODEL);
        gemm(HH, nullptr, Bm, nullptr, nullptr, NNODES, DMODEL, din, din, DMODEL, 0);
        k_node_attn<<<(NNODES * NHEADS + TPB - 1) / TPB, TPB, 0, stream>>>(Bm, asr, adr, AS, AD);
        k_gat_init<<<(NNODES * DMODEL + TPB - 1) / TPB, TPB, 0, stream>>>(C, MS, SS);
        k_gat_edge1<<<(nED + TPB - 1) / TPB, TPB, 0, stream>>>(ei, AS, AD, EW, MS);
        k_gat_edge2<<<(nED + TPB - 1) / TPB, TPB, 0, stream>>>(ei, EW, MS, SS);
        k_gat_edge3<<<ETOT, DMODEL, 0, stream>>>(ei, EW, SS, Bm, C);
        k_gat_final<<<(NNODES * DMODEL + TPB - 1) / TPB, TPB, 0, stream>>>(C, bg, A);
    }

    // ---- decoder layer (A = h_in) ----
    cast16(A, H2, NNODES * DMODEL);
    dim3 fgrid(NNODES / 16, NHEADS);
    // self-attention
    castW(F(21), DMODEL, DMODEL);
    gemm(H2, F(22), nullptr, QH, nullptr, NNODES, DMODEL, DMODEL, DMODEL, DMODEL, 0);
    castW(F(23), DMODEL, DMODEL);
    gemm(H2, F(24), nullptr, KH, nullptr, NNODES, DMODEL, DMODEL, DMODEL, DMODEL, 0);
    castW(F(25), DMODEL, DMODEL);
    gemm(H2, F(26), nullptr, nullptr, VT, NNODES, DMODEL, DMODEL, DMODEL, DMODEL, 0);
    flash_attn_wmma<<<fgrid, 32, 0, stream>>>(QH, KH, VT, C, NNODES, DMODEL);
    cast16(C, HH, NNODES * DMODEL);
    castW(F(27), DMODEL, DMODEL);
    gemm(HH, F(28), Bm, nullptr, nullptr, NNODES, DMODEL, DMODEL, DMODEL, DMODEL, 0);
    k_residual_ln<<<NNODES, DMODEL, 0, stream>>>(A, Bm, F(37), F(38), X1);
    // cross-attention (mem = h_in via H2)
    cast16(X1, HH, NNODES * DMODEL);
    castW(F(29), DMODEL, DMODEL);
    gemm(HH, F(30), nullptr, QH, nullptr, NNODES, DMODEL, DMODEL, DMODEL, DMODEL, 0);
    castW(F(31), DMODEL, DMODEL);
    gemm(H2, F(32), nullptr, KH, nullptr, NNODES, DMODEL, DMODEL, DMODEL, DMODEL, 0);
    castW(F(33), DMODEL, DMODEL);
    gemm(H2, F(34), nullptr, nullptr, VT, NNODES, DMODEL, DMODEL, DMODEL, DMODEL, 0);
    flash_attn_wmma<<<fgrid, 32, 0, stream>>>(QH, KH, VT, C, NNODES, DMODEL);
    cast16(C, HH, NNODES * DMODEL);
    castW(F(35), DMODEL, DMODEL);
    gemm(HH, F(36), Bm, nullptr, nullptr, NNODES, DMODEL, DMODEL, DMODEL, DMODEL, 0);
    k_residual_ln<<<NNODES, DMODEL, 0, stream>>>(X1, Bm, F(39), F(40), A);   // A = x2
    // FFN
    cast16(A, HH, NNODES * DMODEL);
    castW(F(43), DMODEL, FFDIM);
    gemm(HH, F(44), nullptr, TH, nullptr, NNODES, FFDIM, DMODEL, DMODEL, FFDIM, 1);
    castW(F(45), FFDIM, DMODEL);
    gemm(TH, F(46), Bm, nullptr, nullptr, NNODES, DMODEL, FFDIM, FFDIM, DMODEL, 0);
    k_residual_ln<<<NNODES, DMODEL, 0, stream>>>(A, Bm, F(41), F(42), X1);   // final h

    // ---- pool + heads ----
    k_pool_max<<<(NGRAPH * DMODEL + TPB - 1) / TPB, TPB, 0, stream>>>(X1, POOL);
    k_fc<<<(NGRAPH * 128 + TPB - 1) / TPB, TPB, 0, stream>>>(POOL, fc1W, fc1b, Y1,
                                                             NGRAPH, 128, DMODEL, 1);
    k_fc<<<(NGRAPH * OUTDIM + TPB - 1) / TPB, TPB, 0, stream>>>(Y1, fc2W, fc2b, (float*)d_out,
                                                                NGRAPH, OUTDIM, 128, 0);
}
